// T5Gemma2MergedAttention_90855738180359
// MI455X (gfx1250) — matrix operand order repaired and checked by reading, not compiled, not run on hardware
//
#include <hip/hip_runtime.h>
#include <cstdint>
#include <cstddef>

// ---------------- problem constants ----------------
#define B_   4
#define T_   1024
#define E_   1024
#define D_   2048
#define NH_  16      // query heads
#define KH_  8       // kv heads
#define H_   256     // head dim
#define S_   2048    // T + E
#define QKV_COLS 8192  // N*H + K*H + K*H
#define KV_COLS  4096  // K*H + K*H
#define SCALE_    0.0625f            // H^-0.5
#define SOFTCAP_  50.0f
#define NEG_BIG_ (-3.3895313892515355e38f)  // bf16 min

// ---------------- CDNA5 WMMA types ----------------
typedef __attribute__((ext_vector_type(16))) __bf16 bf16x16;
typedef __attribute__((ext_vector_type(8)))  float  f32x8;

union Frag {
  bf16x16 v;
  unsigned int w[8];
};
union U4 {
  uint4 q;
  unsigned short s[8];
  unsigned int w[4];
};
union U2 {
  unsigned long long u;
  unsigned short s[4];
};

__device__ __forceinline__ unsigned short f2bf(float f) {
  union { float f; unsigned u; } x; x.f = f;
  unsigned r = x.u + 0x7fffu + ((x.u >> 16) & 1u);   // RNE
  return (unsigned short)(r >> 16);
}
__device__ __forceinline__ float bf2f(unsigned short h) {
  union { unsigned u; float f; } x; x.u = ((unsigned)h) << 16;
  return x.f;
}
// 16-bit A-matrix 16x32 layout: VGPR j holds K pair {kApair(j)+8*half, +1}
__device__ __forceinline__ int kApair(int j) { return (j < 4) ? 2 * j : 2 * j + 8; }

__device__ __forceinline__ f32x8 wmma_bf16(const Frag& a, const Frag& b, f32x8 c) {
  return __builtin_amdgcn_wmma_f32_16x16x32_bf16(
      false, a.v, false, b.v, (short)0, c, false, false);
}

// CDNA5 async global->LDS copy (ASYNCcnt-tracked; VDST = per-lane LDS byte addr)
__device__ __forceinline__ void async_load_b128(const void* gaddr, void* lds) {
  unsigned laddr = (unsigned)(uintptr_t)lds;                 // LDS byte offset
  unsigned long long ga = (unsigned long long)(uintptr_t)gaddr;
  asm volatile("global_load_async_to_lds_b128 %0, %1, off"
               :: "v"(laddr), "v"(ga) : "memory");
}
__device__ __forceinline__ void wait_async0() {
  asm volatile("s_wait_asynccnt 0" ::: "memory");
}

// ================= conversions =================
__global__ __launch_bounds__(256) void cvt_f32_bf16_kernel(
    const float* __restrict__ src, unsigned short* __restrict__ dst, size_t n4) {
  size_t i = (size_t)blockIdx.x * 256 + threadIdx.x;   // over n/4
  if (i >= n4) return;
  const float4 f = *(const float4*)(src + i * 4);
  U2 o;
  o.s[0] = f2bf(f.x); o.s[1] = f2bf(f.y); o.s[2] = f2bf(f.z); o.s[3] = f2bf(f.w);
  *(unsigned long long*)(dst + i * 4) = o.u;
}

// Build fused (D x 8192) weight: cols [0,4096)=q_w, [4096,6144)=k_w, [6144,8192)=v_w
// q_w/k_w/v_w are (heads, D, H); Wbig[dd][c] = w[n][dd][h]; 4 cols per thread.
__global__ __launch_bounds__(256) void build_wbig_kernel(
    const float* __restrict__ qw, const float* __restrict__ kw,
    const float* __restrict__ vw, unsigned short* __restrict__ wb) {
  size_t idx4 = (size_t)blockIdx.x * 256 + threadIdx.x;   // D_*8192/4 total
  int dd = (int)(idx4 >> 11);
  int c  = (int)(idx4 & 2047) * 4;                        // 4 aligned cols
  const float* src;
  if (c < 4096) {
    int n = c >> 8, h = c & 255;
    src = qw + ((size_t)n * D_ + dd) * H_ + h;
  } else if (c < 6144) {
    int n = (c - 4096) >> 8, h = c & 255;
    src = kw + ((size_t)n * D_ + dd) * H_ + h;
  } else {
    int n = (c - 6144) >> 8, h = c & 255;
    src = vw + ((size_t)n * D_ + dd) * H_ + h;
  }
  const float4 f = *(const float4*)src;
  U2 o;
  o.s[0] = f2bf(f.x); o.s[1] = f2bf(f.y); o.s[2] = f2bf(f.z); o.s[3] = f2bf(f.w);
  *(unsigned long long*)(wb + (size_t)dd * QKV_COLS + c) = o.u;
}

// ================= tiled bf16 WMMA GEMM =================
// C(MxN) = A(MxK) * W(KxN); A row-major bf16, W row-major bf16.
// 64x64 block tile, 8 waves: wave = (ri:2)*(sj:4); each wave -> two 16x16 row
// subtiles at rows 16*(2*ri+p), cols 16*sj. M,N multiples of 64; K multiple of 32.
template <bool OUT_BF16>
__global__ __launch_bounds__(256) void gemm_bf16_wmma(
    const unsigned short* __restrict__ A, int lda,
    const unsigned short* __restrict__ W, int ldw,
    void* __restrict__ Cp, int ldc, int Kd) {
  __shared__ unsigned short As[64][40];  // [row][k], stride 80B (16B multiple)
  __shared__ unsigned short Bs[64][40];  // [col][k] = W^T tile

  const int tid = threadIdx.x;
  const int wv = tid >> 5, lane = tid & 31, m = lane & 15, hh = lane >> 4;
  const int ri = wv >> 2, sj = wv & 3;
  const size_t row0 = (size_t)blockIdx.y * 64;
  const size_t col0 = (size_t)blockIdx.x * 64;

  const int ar = tid >> 2, ac = (tid & 3) * 8;            // A-tile slot (1 uint4)
  const int wk = tid >> 3, wc = (tid & 7) * 8;            // W-tile slot (1 uint4)

  f32x8 acc[2];
#pragma unroll
  for (int p = 0; p < 2; ++p)
#pragma unroll
    for (int v = 0; v < 8; ++v) acc[p][v] = 0.f;

  for (int k0 = 0; k0 < Kd; k0 += 32) {
    // A tile: 64x32, one 128-bit load per thread
    *(uint4*)&As[ar][ac] = *(const uint4*)(A + (row0 + ar) * lda + (k0 + ac));
    // W tile transposed: 128-bit load along columns, u16 scatter into Bs
    {
      U4 d;
      d.q = *(const uint4*)(W + (size_t)(k0 + wk) * ldw + (col0 + wc));
#pragma unroll
      for (int j = 0; j < 8; ++j) Bs[wc + j][wk] = d.s[j];
    }
    if (k0 + 32 < Kd) {   // prefetch next tiles into cache (global_prefetch_b8)
      __builtin_prefetch(A + (row0 + ar) * lda + (k0 + 32 + ac), 0, 3);
      __builtin_prefetch(W + (size_t)(k0 + 32 + wk) * ldw + (col0 + wc), 0, 3);
    }
    __syncthreads();

    Frag bfr;
#pragma unroll
    for (int j = 0; j < 8; ++j)
      bfr.w[j] = *(const unsigned int*)&Bs[16 * sj + m][kApair(j) + 8 * hh];
#pragma unroll
    for (int p = 0; p < 2; ++p) {
      const int rsub = 16 * (2 * ri + p);
      Frag a;
#pragma unroll
      for (int j = 0; j < 8; ++j)
        a.w[j] = *(const unsigned int*)&As[rsub + m][kApair(j) + 8 * hh];
      acc[p] = wmma_bf16(a, bfr, acc[p]);
    }
    __syncthreads();
  }

#pragma unroll
  for (int p = 0; p < 2; ++p) {
#pragma unroll
    for (int v = 0; v < 8; ++v) {
      size_t r = row0 + 16 * (2 * ri + p) + v + 8 * hh;
      size_t c = col0 + 16 * sj + m;
      float val = acc[p][v];
      if (OUT_BF16)
        ((unsigned short*)Cp)[r * (size_t)ldc + c] = f2bf(val);
      else
        ((float*)Cp)[r * (size_t)ldc + c] = val;
    }
  }
}

// ================= RMS-norm / RoPE finalization =================
// One wave32 per H=256 vector; lane owns 8 contiguous h = lane*8..lane*8+7
// (one uint4 load/store). RoPE partner (h, h+128) found via lane xor 16.

__device__ __forceinline__ void rms_rope_vec(
    const unsigned short* __restrict__ src, const float* __restrict__ scale,
    int pos, float outmul, unsigned short* __restrict__ dst, int lane) {
  U4 d;
  d.q = *(const uint4*)(src + lane * 8);
  float x[8];
  float ss = 0.f;
#pragma unroll
  for (int j = 0; j < 8; ++j) {
    x[j] = bf2f(d.s[j]);
    ss += x[j] * x[j];
  }
  for (int off = 16; off; off >>= 1) ss += __shfl_xor(ss, off, 32);
  const float inv = rsqrtf(ss * (1.f / 256.f) + 1e-6f);
#pragma unroll
  for (int j = 0; j < 8; ++j) x[j] = x[j] * inv * (1.f + scale[lane * 8 + j]);
  // RoPE
  const int h1b = (lane & 15) * 8;       // 0..127 base for the pair angle
  const bool hi = lane >= 16;            // this lane holds x2 if true
  U4 o;
#pragma unroll
  for (int j = 0; j < 8; ++j) {
    float ts = __powf(10000.f, (float)(h1b + j) * (1.f / 128.f));
    float ang = (float)pos / ts;
    float sn, cs;
    __sincosf(ang, &sn, &cs);
    float p = __shfl_xor(x[j], 16, 32);  // partner half
    float y = hi ? (x[j] * cs + p * sn)  // x2*cos + x1*sin
                 : (x[j] * cs - p * sn); // x1*cos - x2*sin
    o.s[j] = f2bf(y * outmul);
  }
  *(uint4*)(dst + lane * 8) = o.q;
}

__global__ __launch_bounds__(256) void finalize_q_kernel(
    const unsigned short* __restrict__ qkvs, const int* __restrict__ posids,
    const float* __restrict__ qscale, unsigned short* __restrict__ qb) {
  const int wv = threadIdx.x >> 5, lane = threadIdx.x & 31;
  const int vid = blockIdx.x * 8 + wv;           // < B*T*N
  const int n = vid & 15;
  const int bt = vid >> 4;
  const int b = bt >> 10, t = bt & 1023;
  rms_rope_vec(qkvs + (size_t)bt * QKV_COLS + n * H_, qscale, posids[bt], SCALE_,
               qb + ((size_t)(b * NH_ + n) * T_ + t) * H_, lane);
}

__global__ __launch_bounds__(256) void finalize_selfk_kernel(
    const unsigned short* __restrict__ qkvs, const int* __restrict__ posids,
    const float* __restrict__ kscale, unsigned short* __restrict__ keys) {
  const int wv = threadIdx.x >> 5, lane = threadIdx.x & 31;
  const int vid = blockIdx.x * 8 + wv;           // < B*T*K
  const int k = vid & 7;
  const int bt = vid >> 3;
  const int b = bt >> 10, t = bt & 1023;
  rms_rope_vec(qkvs + (size_t)bt * QKV_COLS + 4096 + k * H_, kscale, posids[bt], 1.f,
               keys + ((size_t)(b * KH_ + k) * S_ + t) * H_, lane);
}

__global__ __launch_bounds__(256) void finalize_crossk_kernel(
    const unsigned short* __restrict__ kvc, const float* __restrict__ kscale,
    unsigned short* __restrict__ keys) {
  const int wv = threadIdx.x >> 5, lane = threadIdx.x & 31;
  const int vid = blockIdx.x * 8 + wv;           // < B*E*K
  const int k = vid & 7;
  const int be = vid >> 3;
  const int b = be >> 10, e = be & 1023;
  const unsigned short* src = kvc + (size_t)be * KV_COLS + k * H_;
  U4 d;
  d.q = *(const uint4*)(src + lane * 8);
  float x[8];
  float ss = 0.f;
#pragma unroll
  for (int j = 0; j < 8; ++j) {
    x[j] = bf2f(d.s[j]);
    ss += x[j] * x[j];
  }
  for (int off = 16; off; off >>= 1) ss += __shfl_xor(ss, off, 32);
  const float inv = rsqrtf(ss * (1.f / 256.f) + 1e-6f);
  U4 o;
#pragma unroll
  for (int j = 0; j < 8; ++j)
    o.s[j] = f2bf(x[j] * inv * (1.f + kscale[lane * 8 + j]));
  unsigned short* dst = keys + ((size_t)(b * KH_ + k) * S_ + (T_ + e)) * H_;
  *(uint4*)(dst + lane * 8) = o.q;
}

__global__ __launch_bounds__(256) void copy_v_self_kernel(
    const unsigned short* __restrict__ qkvs, unsigned short* __restrict__ vals) {
  size_t idx8 = (size_t)blockIdx.x * 256 + threadIdx.x;  // B*T*K*H/8
  int h = (int)(idx8 & 31) * 8;
  int k = (int)((idx8 >> 5) & 7);
  int bt = (int)(idx8 >> 8);
  int b = bt >> 10, t = bt & 1023;
  *(uint4*)&vals[((size_t)(b * KH_ + k) * S_ + t) * H_ + h] =
      *(const uint4*)&qkvs[(size_t)bt * QKV_COLS + 6144 + k * H_ + h];
}

__global__ __launch_bounds__(256) void copy_v_cross_kernel(
    const unsigned short* __restrict__ kvc, unsigned short* __restrict__ vals) {
  size_t idx8 = (size_t)blockIdx.x * 256 + threadIdx.x;  // B*E*K*H/8
  int h = (int)(idx8 & 31) * 8;
  int k = (int)((idx8 >> 5) & 7);
  int be = (int)(idx8 >> 8);
  int b = be >> 10, e = be & 1023;
  *(uint4*)&vals[((size_t)(b * KH_ + k) * S_ + (T_ + e)) * H_ + h] =
      *(const uint4*)&kvc[(size_t)be * KV_COLS + 2048 + k * H_ + h];
}

// ================= flash attention (WMMA QK^T + PV) =================
// Block: (b, kv-head, g, 32 query rows). 8 waves: wave = (qi:2)*(cg:4).
// Logits phase: wave -> 16x16 subtile (rows 16*qi, cols 16*cg) of 32x64 tile.
// PV phase: wave keeps 4 accumulators covering cols 16*(cg*4+cc), cc<4.
// Q and K tiles arrive via async global->LDS B128 copies (ASYNCcnt).
__global__ __launch_bounds__(256) void attn_fa_wmma_kernel(
    const unsigned short* __restrict__ qb,
    const unsigned short* __restrict__ keys,
    const unsigned short* __restrict__ vals,
    const unsigned char* __restrict__ mask,
    unsigned short* __restrict__ attn) {
  __shared__ unsigned short Qs[32][H_ + 8];   // [t][h], 528B stride (16B mult)
  __shared__ unsigned short Ks[64][H_ + 8];   // [s][h]
  __shared__ unsigned short Vt[H_][64 + 4];   // [h][s] (transposed V tile)
  __shared__ float Lg[32][65];                // logits tile
  __shared__ unsigned short Ps[32][68];       // probs (bf16)
  __shared__ float rmax[32], rsum[32], rcorr[32];

  const int b = blockIdx.z;
  const int hk = blockIdx.y >> 1, g = blockIdx.y & 1;
  const int n = hk * 2 + g;                   // query head = k*G + g
  const int t0 = blockIdx.x * 32;
  const int tid = threadIdx.x;
  const int wv = tid >> 5, lane = tid & 31, m = lane & 15, hh = lane >> 4;
  const int qi = wv >> 2, cg = wv & 3;

  // async-load Q tile: 32x256 u16 = 1024 x B128, 4 per thread
  const unsigned short* qsrc = qb + ((size_t)(b * NH_ + n) * T_ + t0) * H_;
#pragma unroll
  for (int i = 0; i < 4; ++i) {
    int u4 = i * 256 + tid;
    int r = u4 >> 5, c = (u4 & 31) * 8;
    async_load_b128(qsrc + (size_t)r * H_ + c, &Qs[r][c]);
  }
  if (tid < 32) { rmax[tid] = NEG_BIG_; rsum[tid] = 0.f; rcorr[tid] = 0.f; }

  f32x8 oacc[4];
#pragma unroll
  for (int cc = 0; cc < 4; ++cc)
#pragma unroll
    for (int v = 0; v < 8; ++v) oacc[cc][v] = 0.f;

  const size_t kvbase = (size_t)(b * KH_ + hk) * S_ * H_;

  for (int s0 = 0; s0 < S_; s0 += 64) {
    // fully-masked causal tile (self region, strictly above diagonal): skip
    if (s0 < T_ && s0 > t0 + 31) continue;
    __syncthreads();   // prior-iteration LDS consumers are done

    const unsigned short* kt = keys + kvbase + (size_t)s0 * H_;
    const unsigned short* vt = vals + kvbase + (size_t)s0 * H_;
    // K tile: 64x256 u16 = 2048 x B128 async copies, 8 per thread
#pragma unroll
    for (int i = 0; i < 8; ++i) {
      int u4 = i * 256 + tid;
      int s = u4 >> 5, c = (u4 & 31) * 8;
      async_load_b128(kt + (size_t)s * H_ + c, &Ks[s][c]);
    }
    // V tile: 128-bit loads, u16 transpose-scatter into Vt
#pragma unroll
    for (int i = 0; i < 8; ++i) {
      int u4 = i * 256 + tid;
      int s = u4 >> 5, c = (u4 & 31) * 8;
      U4 d;
      d.q = *(const uint4*)(vt + (size_t)s * H_ + c);
#pragma unroll
      for (int j = 0; j < 8; ++j) Vt[c + j][s] = d.s[j];
    }
    wait_async0();     // drain ASYNCcnt (covers Q on first live tile + K)
    __syncthreads();

    // ---- logits: Q(32xH) * K^T(Hx64), subtile per wave ----
    f32x8 lacc;
#pragma unroll
    for (int v = 0; v < 8; ++v) lacc[v] = 0.f;
#pragma unroll
    for (int kk = 0; kk < H_; kk += 32) {
      Frag a, bfr;
#pragma unroll
      for (int j = 0; j < 8; ++j)
        a.w[j] = *(const unsigned int*)&Qs[16 * qi + m][kk + kApair(j) + 8 * hh];
#pragma unroll
      for (int j = 0; j < 8; ++j)
        bfr.w[j] = *(const unsigned int*)&Ks[16 * cg + m][kk + kApair(j) + 8 * hh];
      lacc = wmma_bf16(a, bfr, lacc);
    }
#pragma unroll
    for (int v = 0; v < 8; ++v) {
      int r = 16 * qi + v + 8 * hh;
      int c = 16 * cg + m;
      int t = t0 + r;
      int s = s0 + c;
      float z = lacc[v];
      z = SOFTCAP_ * tanhf(z * (1.f / SOFTCAP_));
      if (mask[((size_t)b * T_ + t) * S_ + s] == 0) z = NEG_BIG_;
      Lg[r][c] = z;
    }
    __syncthreads();

    // ---- online softmax: 8 threads per row, shuffle sub-reductions ----
    {
      const int r = tid >> 3;
      const int sub = tid & 7;
      float mx = rmax[r];
      float tm = NEG_BIG_;
#pragma unroll
      for (int c8 = 0; c8 < 8; ++c8) tm = fmaxf(tm, Lg[r][sub * 8 + c8]);
      tm = fmaxf(tm, __shfl_xor(tm, 1, 32));
      tm = fmaxf(tm, __shfl_xor(tm, 2, 32));
      tm = fmaxf(tm, __shfl_xor(tm, 4, 32));
      float nm = fmaxf(mx, tm);
      float corr = __expf(mx - nm);   // 0 on first live tile (underflows)
      float sum = 0.f;
#pragma unroll
      for (int c8 = 0; c8 < 8; ++c8) {
        float p = __expf(Lg[r][sub * 8 + c8] - nm);
        Ps[r][sub * 8 + c8] = f2bf(p);
        sum += p;
      }
      sum += __shfl_xor(sum, 1, 32);
      sum += __shfl_xor(sum, 2, 32);
      sum += __shfl_xor(sum, 4, 32);
      if (sub == 0) {
        rsum[r] = rsum[r] * corr + sum;
        rmax[r] = nm;
        rcorr[r] = corr;
      }
    }
    __syncthreads();

    // ---- PV: O += P(32x64) * V(64xH) ----
#pragma unroll
    for (int cc = 0; cc < 4; ++cc) {
      const int csub = 16 * (cg * 4 + cc);
      f32x8 acc = oacc[cc];
#pragma unroll
      for (int v = 0; v < 8; ++v) acc[v] *= rcorr[16 * qi + v + 8 * hh];
#pragma unroll
      for (int kk = 0; kk < 64; kk += 32) {
        Frag a, bfr;
#pragma unroll
        for (int j = 0; j < 8; ++j)
          a.w[j] = *(const unsigned int*)&Ps[16 * qi + m][kk + kApair(j) + 8 * hh];
#pragma unroll
        for (int j = 0; j < 8; ++j)
          bfr.w[j] = *(const unsigned int*)&Vt[csub + m][kk + kApair(j) + 8 * hh];
        acc = wmma_bf16(a, bfr, acc);
      }
      oacc[cc] = acc;
    }
  }
  __syncthreads();

  // normalize and store attn as (B,T,N,H) bf16 — A matrix of output projection
#pragma unroll
  for (int cc = 0; cc < 4; ++cc) {
#pragma unroll
    for (int v = 0; v < 8; ++v) {
      int r = 16 * qi + v + 8 * hh;
      int t = t0 + r;
      int h = 16 * (cg * 4 + cc) + m;
      float denom = fmaxf(rsum[r], 1e-30f);
      float o = oacc[cc][v] / denom;
      attn[(((size_t)b * T_ + t) * NH_ + n) * H_ + h] = f2bf(o);
    }
  }
}

// ================= host orchestration =================
extern "C" void kernel_launch(void* const* d_in, const int* in_sizes, int n_in,
                              void* d_out, int out_size, void* d_ws, size_t ws_size,
                              hipStream_t stream) {
  (void)in_sizes; (void)n_in; (void)out_size; (void)ws_size;
  const float* hs  = (const float*)d_in[0];           // (B,T,D) f32
  const float* enc = (const float*)d_in[1];           // (B,E,D) f32
  const int*   pos = (const int*)d_in[2];             // (B,T) i32
  const unsigned char* mask = (const unsigned char*)d_in[3]; // (B,1,T,S) bool
  const float* qw = (const float*)d_in[4];            // (N,D,H)
  const float* kw = (const float*)d_in[5];            // (K,D,H)
  const float* vw = (const float*)d_in[6];            // (K,D,H)
  const float* ow = (const float*)d_in[7];            // (N,H,D)
  const float* qs = (const float*)d_in[8];            // (H,)
  const float* ks = (const float*)d_in[9];            // (H,)

  char* wp = (char*)d_ws;
  auto carve = [&](size_t bytes) -> void* {
    void* p = (void*)wp;
    wp += (bytes + 255) & ~(size_t)255;
    return p;
  };
  unsigned short* hsb  = (unsigned short*)carve((size_t)B_ * T_ * D_ * 2);
  unsigned short* encb = (unsigned short*)carve((size_t)B_ * E_ * D_ * 2);
  unsigned short* wbig = (unsigned short*)carve((size_t)D_ * QKV_COLS * 2);
  unsigned short* wo   = (unsigned short*)carve((size_t)NH_ * H_ * D_ * 2);
  unsigned short* qkvs = (unsigned short*)carve((size_t)B_ * T_ * QKV_COLS * 2);
  unsigned short* kvc  = (unsigned short*)carve((size_t)B_ * E_ * KV_COLS * 2);
  unsigned short* qb   = (unsigned short*)carve((size_t)B_ * NH_ * T_ * H_ * 2);
  unsigned short* keys = (unsigned short*)carve((size_t)B_ * KH_ * S_ * H_ * 2);
  unsigned short* vals = (unsigned short*)carve((size_t)B_ * KH_ * S_ * H_ * 2);
  unsigned short* attn = (unsigned short*)carve((size_t)B_ * T_ * NH_ * H_ * 2);

  // stage 1: bf16 conversions / weight fusion (all 128-bit vectorized)
  {
    size_t n4 = (size_t)B_ * T_ * D_ / 4;
    cvt_f32_bf16_kernel<<<(unsigned)((n4 + 255) / 256), 256, 0, stream>>>(hs, hsb, n4);
    n4 = (size_t)B_ * E_ * D_ / 4;
    cvt_f32_bf16_kernel<<<(unsigned)((n4 + 255) / 256), 256, 0, stream>>>(enc, encb, n4);
    n4 = (size_t)NH_ * H_ * D_ / 4;   // o_w (N,H,D) == (N*H x D) row-major
    cvt_f32_bf16_kernel<<<(unsigned)((n4 + 255) / 256), 256, 0, stream>>>(ow, wo, n4);
    build_wbig_kernel<<<(unsigned)(((size_t)D_ * QKV_COLS / 4) / 256), 256, 0, stream>>>(
        qw, kw, vw, wbig);
  }

  // stage 2: projections (bf16 WMMA GEMMs)
  gemm_bf16_wmma<true><<<dim3(QKV_COLS / 64, (B_ * T_) / 64), 256, 0, stream>>>(
      hsb, D_, wbig, QKV_COLS, (void*)qkvs, QKV_COLS, D_);
  gemm_bf16_wmma<true><<<dim3(KV_COLS / 64, (B_ * E_) / 64), 256, 0, stream>>>(
      encb, D_, wbig + 4096, QKV_COLS, (void*)kvc, KV_COLS, D_);

  // stage 3: RMS norm + RoPE + layout
  finalize_q_kernel<<<(B_ * T_ * NH_) / 8, 256, 0, stream>>>(qkvs, pos, qs, qb);
  finalize_selfk_kernel<<<(B_ * T_ * KH_) / 8, 256, 0, stream>>>(qkvs, pos, ks, keys);
  finalize_crossk_kernel<<<(B_ * E_ * KH_) / 8, 256, 0, stream>>>(kvc, ks, keys);
  copy_v_self_kernel<<<(unsigned)(((size_t)B_ * T_ * KH_ * H_ / 8) / 256), 256, 0,
                       stream>>>(qkvs, vals);
  copy_v_cross_kernel<<<(unsigned)(((size_t)B_ * E_ * KH_ * H_ / 8) / 256), 256, 0,
                        stream>>>(kvc, vals);

  // stage 4: flash attention
  attn_fa_wmma_kernel<<<dim3(T_ / 32, KH_ * 2, B_), 256, 0, stream>>>(
      qb, keys, vals, mask, attn);

  // stage 5: output projection -> f32 d_out
  gemm_bf16_wmma<false><<<dim3(D_ / 64, (B_ * T_) / 64), 256, 0, stream>>>(
      attn, NH_ * H_, wo, D_, d_out, D_, NH_ * H_);
}
